// Block_1305670058379
// MI455X (gfx1250) — 16-bit and 8-bit matrix operand order repaired and checked by reading, compiled, not run on hardware
//
#include <hip/hip_runtime.h>
#include <hip/hip_bf16.h>
#include <stdint.h>

typedef int      v8i  __attribute__((ext_vector_type(8)));
typedef float    v8f  __attribute__((ext_vector_type(8)));
typedef _Float16 v16h __attribute__((ext_vector_type(16)));

#define LN_EPS 1e-5f

__device__ __forceinline__ v8i load_a_s8(const char* p, int ld, int lane) {
  const char* row = p + (lane & 15) * ld;
  const int kb = (lane >> 4) << 3;
  v8i a;
#pragma unroll
  for (int v = 0; v < 8; ++v) {
    const int k = ((v >> 1) << 4) + ((v & 1) << 2) + kb;
    a[v] = *(const int*)(row + k);
  }
  return a;
}

__device__ __forceinline__ v8i load_b_s8(const char* p, int ld, int lane) {
  const char* row = p + (lane & 15) * ld;
  const int kb = (lane >> 4) << 3;
  v8i b;
#pragma unroll
  for (int v = 0; v < 8; ++v) {
    const int k = ((v >> 1) << 4) + ((v & 1) << 2) + kb;
    b[v] = *(const int*)(row + k);
  }
  return b;
}

__device__ __forceinline__ v16h load_a_h(const _Float16* p, int ld, int lane) {
  const _Float16* row = p + (size_t)(lane & 15) * ld;
  const int kb = (lane >> 4) << 3;
  v8i t;
#pragma unroll
  for (int v = 0; v < 8; ++v) {
    const int k = ((v & 4) << 2) + ((v & 3) << 1) + kb;
    t[v] = *(const int*)(row + k);
  }
  union { v8i i; v16h h; } u; u.i = t; return u.h;
}

__device__ __forceinline__ v16h load_b_h(const _Float16* p, int ld, int lane) {
  const _Float16* row = p + (size_t)(lane & 15) * ld;
  const int kb = (lane >> 4) << 3;
  v8i t;
#pragma unroll
  for (int v = 0; v < 8; ++v) {
    const int k = ((v & 4) << 2) + ((v & 3) << 1) + kb;
    t[v] = *(const int*)(row + k);
  }
  union { v8i i; v16h h; } u; u.i = t; return u.h;
}

__device__ __forceinline__ uint32_t lds_off(const void* p) {
  return (uint32_t)(uintptr_t)p;
}

__device__ __forceinline__ void async_copy_tile64(uint32_t lds_base, const char* gsrc,
                                                  int ld_bytes, int tid) {
  const int r0 = tid >> 2;
  const int ko = (tid & 3) << 4;
  const uint32_t l0 = lds_base + (uint32_t)(tid << 4);
  const unsigned long long g0 = (unsigned long long)(gsrc + (size_t)r0 * ld_bytes + ko);
  asm volatile("global_load_async_to_lds_b128 %0, %1, off" :: "v"(l0), "v"(g0) : "memory");
  const uint32_t l1 = l0 + 2048;
  const unsigned long long g1 = g0 + (unsigned long long)32 * (unsigned)ld_bytes;
  asm volatile("global_load_async_to_lds_b128 %0, %1, off" :: "v"(l1), "v"(g1) : "memory");
}

__device__ __forceinline__ void wait_async_4() { asm volatile("s_wait_asynccnt 4" ::: "memory"); }
__device__ __forceinline__ void wait_async_2() { asm volatile("s_wait_asynccnt 2" ::: "memory"); }

__global__ void k_ln1_sign(const float* __restrict__ x, const float* __restrict__ g,
                           const float* __restrict__ bt, char* __restrict__ A) {
  __shared__ float s1[256], s2[256];
  const int tid = threadIdx.x;
  const float* xr = x + (size_t)blockIdx.x * 768;
  float vals[3], lsum = 0.f, lsq = 0.f;
#pragma unroll
  for (int i = 0; i < 3; ++i) {
    float v = xr[tid + i * 256];
    vals[i] = v; lsum += v; lsq += v * v;
  }
  s1[tid] = lsum; s2[tid] = lsq; __syncthreads();
  for (int st = 128; st > 0; st >>= 1) {
    if (tid < st) { s1[tid] += s1[tid + st]; s2[tid] += s2[tid + st]; }
    __syncthreads();
  }
  const float mu = s1[0] * (1.f / 768.f);
  const float var = s2[0] * (1.f / 768.f) - mu * mu;
  const float rs = rsqrtf(var + LN_EPS);
  char* Ar = A + (size_t)blockIdx.x * 768;
#pragma unroll
  for (int i = 0; i < 3; ++i) {
    const int c = tid + i * 256;
    const float h = (vals[i] - mu) * rs * g[c] + bt[c];
    Ar[c] = (h >= 0.f) ? (char)1 : (char)-1;
  }
}

__global__ void k_ln2_h(const float* __restrict__ x, const float* __restrict__ g,
                        const float* __restrict__ bt, _Float16* __restrict__ H2) {
  __shared__ float s1[256], s2[256];
  const int tid = threadIdx.x;
  const float* xr = x + (size_t)blockIdx.x * 768;
  float vals[3], lsum = 0.f, lsq = 0.f;
#pragma unroll
  for (int i = 0; i < 3; ++i) {
    float v = xr[tid + i * 256];
    vals[i] = v; lsum += v; lsq += v * v;
  }
  s1[tid] = lsum; s2[tid] = lsq; __syncthreads();
  for (int st = 128; st > 0; st >>= 1) {
    if (tid < st) { s1[tid] += s1[tid + st]; s2[tid] += s2[tid + st]; }
    __syncthreads();
  }
  const float mu = s1[0] * (1.f / 768.f);
  const float var = s2[0] * (1.f / 768.f) - mu * mu;
  const float rs = rsqrtf(var + LN_EPS);
  _Float16* Hr = H2 + (size_t)blockIdx.x * 768;
#pragma unroll
  for (int i = 0; i < 3; ++i) {
    const int c = tid + i * 256;
    Hr[c] = (_Float16)((vals[i] - mu) * rs * g[c] + bt[c]);
  }
}

__global__ void k_prep_sign(const float* __restrict__ w, char* __restrict__ ws,
                            float* __restrict__ alpha, int K) {
  __shared__ float s1[256];
  const int tid = threadIdx.x;
  const float* wr = w + (size_t)blockIdx.x * K;
  float lsum = 0.f;
  for (int c = tid; c < K; c += 256) lsum += fabsf(wr[c]);
  s1[tid] = lsum; __syncthreads();
  for (int st = 128; st > 0; st >>= 1) {
    if (tid < st) s1[tid] += s1[tid + st];
    __syncthreads();
  }
  if (tid == 0) alpha[blockIdx.x] = s1[0] / (float)K;
  char* wsr = ws + (size_t)blockIdx.x * K;
  for (int c = tid; c < K; c += 256) wsr[c] = (wr[c] >= 0.f) ? (char)1 : (char)-1;
}

__global__ void k_prep_h(const float* __restrict__ w, _Float16* __restrict__ wh, int K) {
  __shared__ float s1[256];
  const int tid = threadIdx.x;
  const float* wr = w + (size_t)blockIdx.x * K;
  float lsum = 0.f;
  for (int c = tid; c < K; c += 256) lsum += fabsf(wr[c]);
  s1[tid] = lsum; __syncthreads();
  for (int st = 128; st > 0; st >>= 1) {
    if (tid < st) s1[tid] += s1[tid + st];
    __syncthreads();
  }
  const float al = s1[0] / (float)K;
  _Float16* whr = wh + (size_t)blockIdx.x * K;
  for (int c = tid; c < K; c += 256) whr[c] = (_Float16)((wr[c] >= 0.f) ? al : -al);
}

__global__ void k_gemm_qkv(const char* __restrict__ A, const char* __restrict__ W,
                           char* __restrict__ q, char* __restrict__ kk, char* __restrict__ vt) {
  __shared__ __align__(16) char sb[2][64 * 64];
  const int tid = threadIdx.x;
  const int lane = tid & 31, wave = tid >> 5;
  const int row0 = (blockIdx.y << 6) + (wave << 4);
  const int col0 = blockIdx.x << 6;
  const char* Arow = A + (size_t)row0 * 768;
  const char* Wbase = W + (size_t)col0 * 768;
  const uint32_t lb0 = lds_off(sb[0]), lb1 = lds_off(sb[1]);
  async_copy_tile64(lb0, Wbase, 768, tid);
  v8i a_cur = load_a_s8(Arow, 768, lane);
  v8i acc[4] = {};
  const int NK = 12;
#pragma unroll
  for (int i = 0; i < NK; ++i) {
    const int cur = i & 1;
    const int inext = (i + 1 < NK) ? i + 1 : i;
    async_copy_tile64(cur ? lb0 : lb1, Wbase + inext * 64, 768, tid);
    wait_async_2();
    __syncthreads();
    const v8i a_next = load_a_s8(Arow + inext * 64, 768, lane);
    const char* bs = (const char*)sb[cur];
    v8i bf[4];
#pragma unroll
    for (int j = 0; j < 4; ++j) bf[j] = load_b_s8(bs + (j << 4) * 64, 64, lane);
#pragma unroll
    for (int j = 0; j < 4; ++j)
      acc[j] = __builtin_amdgcn_wmma_i32_16x16x64_iu8(true, a_cur, true, bf[j], acc[j],
                                                      false, false);
    a_cur = a_next;
    __syncthreads();
  }
#pragma unroll
  for (int j = 0; j < 4; ++j) {
    const int o = col0 + (j << 4) + (lane & 15);
    const int which = o / 768;
    const int rem = o - which * 768;
    const int h = rem >> 6, d = rem & 63;
#pragma unroll
    for (int r = 0; r < 8; ++r) {
      const int row = row0 + r + ((lane >> 4) << 3);
      const int bb = row >> 10, n = row & 1023;
      const int bh = bb * 12 + h;
      const char s = (acc[j][r] >= 0) ? (char)1 : (char)-1;
      if (which == 0)      q [(size_t)((bh << 10) + n) * 64 + d] = s;
      else if (which == 1) kk[(size_t)((bh << 10) + n) * 64 + d] = s;
      else                 vt[(size_t)((bh << 6) + d) * 1024 + n] = s;
    }
  }
}

__global__ void k_attn(const char* __restrict__ q, const char* __restrict__ kk,
                       const char* __restrict__ vt, char* __restrict__ So) {
  __shared__ __align__(16) char kt[2][64 * 64];
  __shared__ __align__(16) char vs[2][64 * 64];
  __shared__ __align__(16) char abin[4][16 * 64];
  const int tid = threadIdx.x;
  const int lane = tid & 31, wave = tid >> 5;
  const int bh = blockIdx.y;
  const int row0 = (blockIdx.x << 6) + (wave << 4);
  const v8i aq = load_a_s8(q + (size_t)((bh << 10) + row0) * 64, 64, lane);
  const uint32_t lk0 = lds_off(kt[0]), lk1 = lds_off(kt[1]);
  const uint32_t lv0 = lds_off(vs[0]), lv1 = lds_off(vs[1]);
  const char* ksrc = kk + (size_t)(bh << 10) * 64;
  const char* vsrc = vt + (size_t)(bh << 6) * 1024;
  async_copy_tile64(lk0, ksrc, 64, tid);
  async_copy_tile64(lv0, vsrc, 1024, tid);
  v8i oacc[4] = {};
  const v8i zero = {};
#pragma unroll 2
  for (int ci = 0; ci < 16; ++ci) {
    const int cur = ci & 1;
    const int cnext = ((ci + 1 < 16) ? ci + 1 : ci) << 6;
    async_copy_tile64(cur ? lk0 : lk1, ksrc + (size_t)cnext * 64, 64, tid);
    async_copy_tile64(cur ? lv0 : lv1, vsrc + cnext, 1024, tid);
    wait_async_4();
    __syncthreads();
    const char* ktc = (const char*)kt[cur];
    const char* vsc = (const char*)vs[cur];
    v8i bk[4], t[4], bv[4];
#pragma unroll
    for (int j = 0; j < 4; ++j) bk[j] = load_b_s8(ktc + (j << 4) * 64, 64, lane);
#pragma unroll
    for (int j = 0; j < 4; ++j)
      t[j] = __builtin_amdgcn_wmma_i32_16x16x64_iu8(true, aq, true, bk[j], zero, false, false);
#pragma unroll
    for (int j = 0; j < 4; ++j) {
      const int n = (j << 4) + (lane & 15);
      const int mb = (lane >> 4) << 3;
#pragma unroll
      for (int r = 0; r < 8; ++r)
        abin[wave][(mb + r) * 64 + n] = (t[j][r] > 0) ? (char)1 : (char)0;
    }
    const v8i ab = load_a_s8((const char*)abin[wave], 64, lane);
#pragma unroll
    for (int j = 0; j < 4; ++j) bv[j] = load_b_s8(vsc + (j << 4) * 64, 64, lane);
#pragma unroll
    for (int j = 0; j < 4; ++j)
      oacc[j] = __builtin_amdgcn_wmma_i32_16x16x64_iu8(false, ab, true, bv[j], oacc[j],
                                                       false, false);
    __syncthreads();
  }
  const int bb = bh / 12, h = bh - bb * 12;
#pragma unroll
  for (int j = 0; j < 4; ++j) {
    const int d = (j << 4) + (lane & 15);
#pragma unroll
    for (int r = 0; r < 8; ++r) {
      const int m = row0 + r + ((lane >> 4) << 3);
      So[(size_t)((bb << 10) + m) * 768 + (h << 6) + d] = (oacc[j][r] >= 0) ? (char)1 : (char)-1;
    }
  }
}

__global__ void k_gemm_proj(const char* __restrict__ So, const char* __restrict__ W,
                            const float* __restrict__ alpha, const float* __restrict__ bproj,
                            const float* __restrict__ ls1, const float* __restrict__ x,
                            float* __restrict__ x1) {
  __shared__ __align__(16) char sb[2][64 * 64];
  const int tid = threadIdx.x;
  const int lane = tid & 31, wave = tid >> 5;
  const int row0 = (blockIdx.y << 6) + (wave << 4);
  const int col0 = blockIdx.x << 6;
  const char* Arow = So + (size_t)row0 * 768;
  const char* Wbase = W + (size_t)col0 * 768;
  const uint32_t lb0 = lds_off(sb[0]), lb1 = lds_off(sb[1]);
  async_copy_tile64(lb0, Wbase, 768, tid);
  v8i a_cur = load_a_s8(Arow, 768, lane);
  v8i acc[4] = {};
  const int NK = 12;
#pragma unroll
  for (int i = 0; i < NK; ++i) {
    const int cur = i & 1;
    const int inext = (i + 1 < NK) ? i + 1 : i;
    async_copy_tile64(cur ? lb0 : lb1, Wbase + inext * 64, 768, tid);
    wait_async_2();
    __syncthreads();
    const v8i a_next = load_a_s8(Arow + inext * 64, 768, lane);
    const char* bs = (const char*)sb[cur];
    v8i bf[4];
#pragma unroll
    for (int j = 0; j < 4; ++j) bf[j] = load_b_s8(bs + (j << 4) * 64, 64, lane);
#pragma unroll
    for (int j = 0; j < 4; ++j)
      acc[j] = __builtin_amdgcn_wmma_i32_16x16x64_iu8(true, a_cur, true, bf[j], acc[j],
                                                      false, false);
    a_cur = a_next;
    __syncthreads();
  }
#pragma unroll
  for (int j = 0; j < 4; ++j) {
    const int col = col0 + (j << 4) + (lane & 15);
    const float al = alpha[col], bp = bproj[col], l1 = ls1[col];
#pragma unroll
    for (int r = 0; r < 8; ++r) {
      const int row = row0 + r + ((lane >> 4) << 3);
      const size_t idx = (size_t)row * 768 + col;
      x1[idx] = x[idx] + l1 * fmaf(al, (float)acc[j][r], bp);
    }
  }
}

__global__ void k_fc1(const _Float16* __restrict__ H2, const _Float16* __restrict__ W1,
                      const float* __restrict__ b1, _Float16* __restrict__ M) {
  __shared__ __align__(16) char sb[2][64 * 64];
  const int tid = threadIdx.x;
  const int lane = tid & 31, wave = tid >> 5;
  const int row0 = (blockIdx.y << 6) + (wave << 4);
  const int col0 = blockIdx.x << 6;
  const _Float16* Arow = H2 + (size_t)row0 * 768;
  const char* Wbase = (const char*)(W1 + (size_t)col0 * 768);
  const uint32_t lb0 = lds_off(sb[0]), lb1 = lds_off(sb[1]);
  async_copy_tile64(lb0, Wbase, 768 * 2, tid);
  v16h a_cur = load_a_h(Arow, 768, lane);
  v8f acc[4] = {};
  const int NK = 24;
#pragma unroll
  for (int i = 0; i < NK; ++i) {
    const int cur = i & 1;
    const int inext = (i + 1 < NK) ? i + 1 : i;
    async_copy_tile64(cur ? lb0 : lb1, Wbase + inext * 64, 768 * 2, tid);
    wait_async_2();
    __syncthreads();
    const v16h a_next = load_a_h(Arow + inext * 32, 768, lane);
    const _Float16* bs = (const _Float16*)sb[cur];
    v16h bf[4];
#pragma unroll
    for (int j = 0; j < 4; ++j) bf[j] = load_b_h(bs + (j << 4) * 32, 32, lane);
#pragma unroll
    for (int j = 0; j < 4; ++j)
      acc[j] = __builtin_amdgcn_wmma_f32_16x16x32_f16(false, a_cur, false, bf[j], (short)0,
                                                      acc[j], false, false);
    a_cur = a_next;
    __syncthreads();
  }
#pragma unroll
  for (int j = 0; j < 4; ++j) {
    const int col = col0 + (j << 4) + (lane & 15);
    const float bb = b1[col];
#pragma unroll
    for (int r = 0; r < 8; ++r) {
      const int row = row0 + r + ((lane >> 4) << 3);
      const float z = acc[j][r] + bb;
      const float g = 0.5f * z * (1.0f + erff(z * 0.70710678118654752f));
      M[(size_t)row * 3072 + col] = (_Float16)g;
    }
  }
}

__global__ void k_fc2(const _Float16* __restrict__ M, const _Float16* __restrict__ W2,
                      const float* __restrict__ b2, const float* __restrict__ ls2,
                      const float* __restrict__ x1, float* __restrict__ out) {
  __shared__ __align__(16) char sb[2][64 * 64];
  const int tid = threadIdx.x;
  const int lane = tid & 31, wave = tid >> 5;
  const int row0 = (blockIdx.y << 6) + (wave << 4);
  const int col0 = blockIdx.x << 6;
  const _Float16* Arow = M + (size_t)row0 * 3072;
  const char* Wbase = (const char*)(W2 + (size_t)col0 * 3072);
  const uint32_t lb0 = lds_off(sb[0]), lb1 = lds_off(sb[1]);
  async_copy_tile64(lb0, Wbase, 3072 * 2, tid);
  v16h a_cur = load_a_h(Arow, 3072, lane);
  v8f acc[4] = {};
  const int NK = 96;
#pragma unroll 2
  for (int i = 0; i < NK; ++i) {
    const int cur = i & 1;
    const int inext = (i + 1 < NK) ? i + 1 : i;
    async_copy_tile64(cur ? lb0 : lb1, Wbase + inext * 64, 3072 * 2, tid);
    wait_async_2();
    __syncthreads();
    const v16h a_next = load_a_h(Arow + inext * 32, 3072, lane);
    const _Float16* bs = (const _Float16*)sb[cur];
    v16h bf[4];
#pragma unroll
    for (int j = 0; j < 4; ++j) bf[j] = load_b_h(bs + (j << 4) * 32, 32, lane);
#pragma unroll
    for (int j = 0; j < 4; ++j)
      acc[j] = __builtin_amdgcn_wmma_f32_16x16x32_f16(false, a_cur, false, bf[j], (short)0,
                                                      acc[j], false, false);
    a_cur = a_next;
    __syncthreads();
  }
#pragma unroll
  for (int j = 0; j < 4; ++j) {
    const int col = col0 + (j << 4) + (lane & 15);
    const float bb = b2[col], l2 = ls2[col];
#pragma unroll
    for (int r = 0; r < 8; ++r) {
      const int row = row0 + r + ((lane >> 4) << 3);
      const size_t idx = (size_t)row * 768 + col;
      out[idx] = x1[idx] + l2 * (acc[j][r] + bb);
    }
  }
}

extern "C" void kernel_launch(void* const* d_in, const int* in_sizes, int n_in,
                              void* d_out, int out_size, void* d_ws, size_t ws_size,
                              hipStream_t stream) {
  (void)in_sizes; (void)n_in; (void)out_size; (void)ws_size;
  const float* x      = (const float*)d_in[0];
  const float* ln1_g  = (const float*)d_in[1];
  const float* ln1_b  = (const float*)d_in[2];
  const float* w_qkv  = (const float*)d_in[3];
  const float* w_proj = (const float*)d_in[4];
  const float* b_proj = (const float*)d_in[5];
  const float* ls1_g  = (const float*)d_in[6];
  const float* ln2_g  = (const float*)d_in[7];
  const float* ln2_b  = (const float*)d_in[8];
  const float* w_fc1  = (const float*)d_in[9];
  const float* b_fc1  = (const float*)d_in[10];
  const float* w_fc2  = (const float*)d_in[11];
  const float* b_fc2  = (const float*)d_in[12];
  const float* ls2_g  = (const float*)d_in[13];
  float* out = (float*)d_out;

  char* ws = (char*)d_ws;
  size_t off = 0;
  auto carve = [&](size_t bytes) -> char* {
    char* p = ws + off;
    off = (off + bytes + 255) & ~(size_t)255;
    return p;
  };
  const size_t BN = 8192;
  char*      A_s8  = carve(BN * 768);
  char*      Wq_s8 = carve(2304 * 768);
  float*     al_q  = (float*)carve(2304 * 4);
  char*      Wp_s8 = carve(768 * 768);
  float*     al_p  = (float*)carve(768 * 4);
  char*      q_s8  = carve((size_t)96 * 1024 * 64);
  char*      k_s8  = carve((size_t)96 * 1024 * 64);
  char*      vt_s8 = carve((size_t)96 * 64 * 1024);
  char*      So_s8 = carve(BN * 768);
  float*     x1    = (float*)carve(BN * 768 * 4);
  _Float16*  h2    = (_Float16*)carve(BN * 768 * 2);
  _Float16*  W1h   = (_Float16*)carve((size_t)3072 * 768 * 2);
  _Float16*  W2h   = (_Float16*)carve((size_t)768 * 3072 * 2);
  _Float16*  Mh    = (_Float16*)carve(BN * 3072 * 2);

  k_ln1_sign <<<8192, 256, 0, stream>>>(x, ln1_g, ln1_b, A_s8);
  k_prep_sign<<<2304, 256, 0, stream>>>(w_qkv, Wq_s8, al_q, 768);
  k_prep_sign<<< 768, 256, 0, stream>>>(w_proj, Wp_s8, al_p, 768);
  k_prep_h   <<<3072, 256, 0, stream>>>(w_fc1, W1h, 768);
  k_prep_h   <<< 768, 256, 0, stream>>>(w_fc2, W2h, 3072);

  k_gemm_qkv <<<dim3(36, 128), 128, 0, stream>>>(A_s8, Wq_s8, q_s8, k_s8, vt_s8);
  k_attn     <<<dim3(16, 96),  128, 0, stream>>>(q_s8, k_s8, vt_s8, So_s8);
  k_gemm_proj<<<dim3(12, 128), 128, 0, stream>>>(So_s8, Wp_s8, al_p, b_proj, ls1_g, x, x1);

  k_ln2_h    <<<8192, 256, 0, stream>>>(x1, ln2_g, ln2_b, h2);
  k_fc1      <<<dim3(48, 128), 128, 0, stream>>>(h2, W1h, b_fc1, Mh);
  k_fc2      <<<dim3(12, 128), 128, 0, stream>>>(Mh, W2h, b_fc2, ls2_g, x1, out);
}
